// QuantDense_39762807226775
// MI455X (gfx1250) — compile-verified, matrix operand order checked by reading, not run on hardware
//
#include <hip/hip_runtime.h>

// ---------------------------------------------------------------------------
// QuantDense forward: Y[M,N] = X[M,K] @ W[K,N] + bias[N]
//   M = 16384, K = 1024, N = 4096, fp32 in memory.
//
// Two-pass bf16x3 split GEMM for MI455X (gfx1250):
//   Pass 1: split X -> Xh/Xl (bf16, [m][k]); split+transpose W -> Wth/Wtl
//           (bf16, [n][k]) into d_ws.  ~160MB traffic, ~7us; results L2-resident.
//   Pass 2: GEMM on V_WMMA_F32_16X16X32_BF16, 3 products per tile
//           (al*bh + ah*bl + ah*bh) ~ fp32 accuracy at 3/8 fp32-WMMA cost.
//           Staging uses global_load_async_to_lds_b128 (ASYNCcnt-tracked),
//           so the hot loop is WMMA/DS-bound instead of VALU-bound.
// ---------------------------------------------------------------------------

typedef __bf16 v16bf __attribute__((ext_vector_type(16)));
typedef float  v8f   __attribute__((ext_vector_type(8)));

union FragAB { uint4 u[2]; v16bf v; };

constexpr int M_TOT = 4 * 4096;   // 16384
constexpr int K_TOT = 1024;
constexpr int N_TOT = 4096;

// ---- pass-2 GEMM tiling ----
constexpr int BM = 128, BN = 128, BK2 = 64;
constexpr int LDE  = 72;             // LDS row stride in bf16 elems (144 B: 64 data + 8 pad)
constexpr int KT2  = K_TOT / BK2;    // 16 stages

// ---- scratch layout (elements of unsigned short) ----
constexpr size_t XH_ELEMS = (size_t)M_TOT * K_TOT;        // 16.7M
constexpr size_t WT_ELEMS = (size_t)N_TOT * K_TOT;        // 4.2M
constexpr size_t WS_NEED  = (2 * XH_ELEMS + 2 * WT_ELEMS) * sizeof(unsigned short); // 80 MB

#if defined(__HIP_DEVICE_COMPILE__) && __has_builtin(__builtin_amdgcn_global_load_async_to_lds_b128)
#define QD_ASYNC 1
#else
#define QD_ASYNC 0
#endif

__device__ __forceinline__ void splitbf(float x, unsigned short& h, unsigned short& l) {
  __bf16 hb = (__bf16)x;             // RNE high part
  float   r = x - (float)hb;         // exact residual in fp32
  __bf16 lb = (__bf16)r;             // RNE low part
  h = __builtin_bit_cast(unsigned short, hb);
  l = __builtin_bit_cast(unsigned short, lb);
}

__device__ __forceinline__ void split4(float4 v, uint2& hp, uint2& lp) {
  unsigned short h[4], l[4];
  splitbf(v.x, h[0], l[0]);
  splitbf(v.y, h[1], l[1]);
  splitbf(v.z, h[2], l[2]);
  splitbf(v.w, h[3], l[3]);
  hp = make_uint2((unsigned)h[0] | ((unsigned)h[1] << 16),
                  (unsigned)h[2] | ((unsigned)h[3] << 16));
  lp = make_uint2((unsigned)l[0] | ((unsigned)l[1] << 16),
                  (unsigned)l[2] | ((unsigned)l[3] << 16));
}

#if QD_ASYNC
// Builtin signature (from hipcc diagnostic): param0 = AS1 pointer to
// __attribute__((vector_size(16))) int, param1 = LDS pointer, then imm offset,
// imm cpol.
typedef int qd_v4i __attribute__((vector_size(4 * sizeof(int))));
typedef __attribute__((address_space(1))) qd_v4i glb_v4i;
typedef __attribute__((address_space(3))) qd_v4i lds_v4i;

__device__ __forceinline__ void async_cp16(const void* g, void* l) {
  __builtin_amdgcn_global_load_async_to_lds_b128((glb_v4i*)g, (lds_v4i*)l, 0, 0);
}
__device__ __forceinline__ void wait_async0() {
#if __has_builtin(__builtin_amdgcn_s_wait_asynccnt)
  __builtin_amdgcn_s_wait_asynccnt(0);
#else
  asm volatile("s_wait_asynccnt 0x0" ::: "memory");
#endif
}
#endif

// ===========================================================================
// Pass 1a: elementwise split of X (8 floats / thread)
// ===========================================================================
__global__ __launch_bounds__(256)
void split_x_kernel(const float* __restrict__ src,
                    unsigned short* __restrict__ hdst,
                    unsigned short* __restrict__ ldst, int n8) {
  int i = blockIdx.x * 256 + threadIdx.x;
  if (i >= n8) return;
  const float4* s = (const float4*)src + (size_t)i * 2;
  float4 a = s[0], b = s[1];
  uint2 h0, l0, h1, l1;
  split4(a, h0, l0);
  split4(b, h1, l1);
  ((uint4*)hdst)[i] = make_uint4(h0.x, h0.y, h1.x, h1.y);
  ((uint4*)ldst)[i] = make_uint4(l0.x, l0.y, l1.x, l1.y);
}

// ===========================================================================
// Pass 1b: split + transpose W[k][n] -> Wt[n][k] (64x64 tiles via LDS)
// ===========================================================================
__global__ __launch_bounds__(256)
void split_wt_kernel(const float* __restrict__ W,
                     unsigned short* __restrict__ Wth,
                     unsigned short* __restrict__ Wtl) {
  constexpr int TLD = 80;                        // padded LDS row stride (elems)
  __shared__ __align__(16) unsigned short th[64 * TLD];
  __shared__ __align__(16) unsigned short tl[64 * TLD];

  const int tid = threadIdx.x;
  const int k0  = blockIdx.x * 64;
  const int n0  = blockIdx.y * 64;

  // phase 1: coalesced read of 64x64 fp32 tile, split, transpose-scatter to LDS
  #pragma unroll
  for (int i = 0; i < 16; ++i) {
    const int idx = tid + i * 256;               // 0..4095
    const int kr  = idx >> 6;                    // k within tile
    const int nc  = idx & 63;                    // n within tile
    const float x = W[(size_t)(k0 + kr) * N_TOT + n0 + nc];
    unsigned short h, l;
    splitbf(x, h, l);
    th[nc * TLD + kr] = h;
    tl[nc * TLD + kr] = l;
  }
  __syncthreads();

  // phase 2: coalesced uint4 writes of Wt rows
  #pragma unroll
  for (int i = 0; i < 2; ++i) {
    const int ch = tid + i * 256;                // 0..511
    const int r  = ch >> 3;                      // n within tile
    const int c8 = ch & 7;                       // 8-elem chunk within row
    uint4 vh = *(const uint4*)&th[r * TLD + c8 * 8];
    uint4 vl = *(const uint4*)&tl[r * TLD + c8 * 8];
    *(uint4*)(Wth + (size_t)(n0 + r) * K_TOT + k0 + c8 * 8) = vh;
    *(uint4*)(Wtl + (size_t)(n0 + r) * K_TOT + k0 + c8 * 8) = vl;
  }
}

// ===========================================================================
// Pass 2: bf16x3 WMMA GEMM, pre-converted operands, async staging
// ===========================================================================
__global__ __launch_bounds__(256)
void qdense_gemm_wmma(const unsigned short* __restrict__ Xh,
                      const unsigned short* __restrict__ Xl,
                      const unsigned short* __restrict__ Wth,
                      const unsigned short* __restrict__ Wtl,
                      const float* __restrict__ Bias,
                      float* __restrict__ Y) {
  // 4 planes x 2 buffers, 128 rows x 72 elems (144 B, 16B-aligned rows)
  __shared__ __align__(16) unsigned short sAh[2][BM * LDE];
  __shared__ __align__(16) unsigned short sAl[2][BM * LDE];
  __shared__ __align__(16) unsigned short sBh[2][BN * LDE];
  __shared__ __align__(16) unsigned short sBl[2][BN * LDE];

  const int tid   = threadIdx.x;
  const int lane  = tid & 31;
  const int wave  = tid >> 5;
  const int wm    = wave >> 1;       // 0..3 : 32-row wave strip
  const int wn    = wave & 1;        // 0..1 : 64-col wave strip
  const int lrow  = lane & 15;
  const int lhalf = lane >> 4;

  const int m0 = blockIdx.y * BM;
  const int n0 = blockIdx.x * BN;

  v8f acc[2][4];
  #pragma unroll
  for (int i = 0; i < 2; ++i)
    #pragma unroll
    for (int j = 0; j < 4; ++j)
      acc[i][j] = v8f{};

  // Each stage moves 4 planes x 128 rows x 128B. 1024 16B-chunks per plane,
  // 4 per thread: chunk = tid + i*256 -> row = chunk>>3, c8 = chunk&7.
  auto stage = [&](int kt, int buf) {
    const int k0 = kt * BK2;
    #pragma unroll
    for (int i = 0; i < 4; ++i) {
      const int ch  = tid + i * 256;
      const int row = ch >> 3;
      const int c8  = ch & 7;
      const size_t goffA = (size_t)(m0 + row) * K_TOT + k0 + c8 * 8;
      const size_t goffB = (size_t)(n0 + row) * K_TOT + k0 + c8 * 8;
      const int    loff  = row * LDE + c8 * 8;
#if QD_ASYNC
      async_cp16(Xh  + goffA, &sAh[buf][loff]);
      async_cp16(Xl  + goffA, &sAl[buf][loff]);
      async_cp16(Wth + goffB, &sBh[buf][loff]);
      async_cp16(Wtl + goffB, &sBl[buf][loff]);
#else
      uint4 a = *(const uint4*)(Xh + goffA);
      uint4 b = *(const uint4*)(Xl + goffA);
      uint4 c = *(const uint4*)(Wth + goffB);
      uint4 d = *(const uint4*)(Wtl + goffB);
      *(uint4*)&sAh[buf][loff] = a;
      *(uint4*)&sAl[buf][loff] = b;
      *(uint4*)&sBh[buf][loff] = c;
      *(uint4*)&sBl[buf][loff] = d;
#endif
    }
  };

  auto compute = [&](int buf) {
    #pragma unroll
    for (int s = 0; s < 2; ++s) {              // two 32-wide k-slabs per stage
      FragAB ah[2], al[2], bh[4], bl[4];
      // 16-bit A 16x32 layout: lanes 0-15 hold K 0-7 & 16-23 (+half*8 for 16-31)
      #pragma unroll
      for (int tm = 0; tm < 2; ++tm) {
        const int r   = wm * 32 + tm * 16 + lrow;
        const int off = r * LDE + s * 32 + lhalf * 8;
        const uint4* ph = (const uint4*)&sAh[buf][off];
        ah[tm].u[0] = ph[0]; ah[tm].u[1] = ph[2];
        const uint4* pl = (const uint4*)&sAl[buf][off];
        al[tm].u[0] = pl[0]; al[tm].u[1] = pl[2];
      }
      #pragma unroll
      for (int tn = 0; tn < 4; ++tn) {
        const int c   = wn * 64 + tn * 16 + lrow;
        const int off = c * LDE + s * 32 + lhalf * 8;
        const uint4* ph = (const uint4*)&sBh[buf][off];
        bh[tn].u[0] = ph[0]; bh[tn].u[1] = ph[2];
        const uint4* pl = (const uint4*)&sBl[buf][off];
        bl[tn].u[0] = pl[0]; bl[tn].u[1] = pl[2];
      }
      #pragma unroll
      for (int tm = 0; tm < 2; ++tm)
        #pragma unroll
        for (int tn = 0; tn < 4; ++tn) {
          v8f cc = acc[tm][tn];
          cc = __builtin_amdgcn_wmma_f32_16x16x32_bf16(false, al[tm].v, false, bh[tn].v,
                                                       (short)0, cc, false, false);
          cc = __builtin_amdgcn_wmma_f32_16x16x32_bf16(false, ah[tm].v, false, bl[tn].v,
                                                       (short)0, cc, false, false);
          cc = __builtin_amdgcn_wmma_f32_16x16x32_bf16(false, ah[tm].v, false, bh[tn].v,
                                                       (short)0, cc, false, false);
          acc[tm][tn] = cc;
        }
    }
  };

  stage(0, 0);
#if QD_ASYNC
  wait_async0();
#endif
  __syncthreads();

  for (int kt = 0; kt < KT2; ++kt) {
    const int cur = kt & 1;
    if (kt + 1 < KT2) stage(kt + 1, cur ^ 1);  // copies in flight behind WMMAs
    compute(cur);
#if QD_ASYNC
    wait_async0();                             // my copies into nxt are done
#endif
    __syncthreads();                           // everyone's copies are done
  }

  // Epilogue: C layout — VGPR r: (M = r + 8*half, N = lane%16).
  #pragma unroll
  for (int tm = 0; tm < 2; ++tm) {
    const int mb = m0 + wm * 32 + tm * 16 + lhalf * 8;
    #pragma unroll
    for (int tn = 0; tn < 4; ++tn) {
      const int n = n0 + wn * 64 + tn * 16 + lrow;
      const float bv = Bias[n];
      #pragma unroll
      for (int r = 0; r < 8; ++r)
        Y[(size_t)(mb + r) * N_TOT + n] = acc[tm][tn][r] + bv;
    }
  }
}

// ===========================================================================
// Fallback: fused single-pass kernel (round-1 version) if ws is too small
// ===========================================================================
constexpr int FLDT = 40;
constexpr int FKT  = K_TOT / 32;

__global__ __launch_bounds__(256)
void qdense_fused_wmma(const float* __restrict__ X, const float* __restrict__ W,
                       const float* __restrict__ Bias, float* __restrict__ Y) {
  __shared__ __align__(16) unsigned short sAh[2][BM * FLDT];
  __shared__ __align__(16) unsigned short sAl[2][BM * FLDT];
  __shared__ __align__(16) unsigned short sBh[2][BN * FLDT];
  __shared__ __align__(16) unsigned short sBl[2][BN * FLDT];

  const int tid = threadIdx.x, lane = tid & 31, wave = tid >> 5;
  const int wm = wave >> 1, wn = wave & 1, lrow = lane & 15, lhalf = lane >> 4;
  const int m0 = blockIdx.y * BM, n0 = blockIdx.x * BN;

  v8f acc[2][4];
  #pragma unroll
  for (int i = 0; i < 2; ++i)
    #pragma unroll
    for (int j = 0; j < 4; ++j) acc[i][j] = v8f{};

  float4 ra[4], rb[4];

  auto load_global = [&](int kt) {
    const int k0 = kt * 32;
    #pragma unroll
    for (int i = 0; i < 4; ++i) {
      const int idx = tid + i * 256;
      ra[i] = *(const float4*)(X + (size_t)(m0 + (idx >> 3)) * K_TOT + k0 + (idx & 7) * 4);
      rb[i] = *(const float4*)(W + (size_t)(k0 + (idx >> 5)) * N_TOT + n0 + (idx & 31) * 4);
    }
  };
  auto store_lds = [&](int buf) {
    #pragma unroll
    for (int i = 0; i < 4; ++i) {
      const int idx = tid + i * 256;
      {
        const int arow = idx >> 3, ac4 = idx & 7;
        uint2 hp, lp;
        split4(ra[i], hp, lp);
        *(uint2*)&sAh[buf][arow * FLDT + ac4 * 4] = hp;
        *(uint2*)&sAl[buf][arow * FLDT + ac4 * 4] = lp;
      }
      {
        const int brow = idx >> 5, bc4 = idx & 31;
        unsigned short h[4], l[4];
        splitbf(rb[i].x, h[0], l[0]);
        splitbf(rb[i].y, h[1], l[1]);
        splitbf(rb[i].z, h[2], l[2]);
        splitbf(rb[i].w, h[3], l[3]);
        #pragma unroll
        for (int j = 0; j < 4; ++j) {
          sBh[buf][(bc4 * 4 + j) * FLDT + brow] = h[j];
          sBl[buf][(bc4 * 4 + j) * FLDT + brow] = l[j];
        }
      }
    }
  };
  auto compute = [&](int buf) {
    FragAB ah[2], al[2], bh[4], bl[4];
    #pragma unroll
    for (int tm = 0; tm < 2; ++tm) {
      const int r = wm * 32 + tm * 16 + lrow;
      const uint4* ph = (const uint4*)&sAh[buf][r * FLDT + lhalf * 8];
      ah[tm].u[0] = ph[0]; ah[tm].u[1] = ph[2];
      const uint4* pl = (const uint4*)&sAl[buf][r * FLDT + lhalf * 8];
      al[tm].u[0] = pl[0]; al[tm].u[1] = pl[2];
    }
    #pragma unroll
    for (int tn = 0; tn < 4; ++tn) {
      const int c = wn * 64 + tn * 16 + lrow;
      const uint4* ph = (const uint4*)&sBh[buf][c * FLDT + lhalf * 8];
      bh[tn].u[0] = ph[0]; bh[tn].u[1] = ph[2];
      const uint4* pl = (const uint4*)&sBl[buf][c * FLDT + lhalf * 8];
      bl[tn].u[0] = pl[0]; bl[tn].u[1] = pl[2];
    }
    #pragma unroll
    for (int tm = 0; tm < 2; ++tm)
      #pragma unroll
      for (int tn = 0; tn < 4; ++tn) {
        v8f cc = acc[tm][tn];
        cc = __builtin_amdgcn_wmma_f32_16x16x32_bf16(false, al[tm].v, false, bh[tn].v,
                                                     (short)0, cc, false, false);
        cc = __builtin_amdgcn_wmma_f32_16x16x32_bf16(false, ah[tm].v, false, bl[tn].v,
                                                     (short)0, cc, false, false);
        cc = __builtin_amdgcn_wmma_f32_16x16x32_bf16(false, ah[tm].v, false, bh[tn].v,
                                                     (short)0, cc, false, false);
        acc[tm][tn] = cc;
      }
  };

  load_global(0);
  store_lds(0);
  __syncthreads();
  for (int kt = 0; kt < FKT; ++kt) {
    const int cur = kt & 1;
    if (kt + 1 < FKT) load_global(kt + 1);
    compute(cur);
    if (kt + 1 < FKT) store_lds(cur ^ 1);
    __syncthreads();
  }

  #pragma unroll
  for (int tm = 0; tm < 2; ++tm) {
    const int mb = m0 + wm * 32 + tm * 16 + lhalf * 8;
    #pragma unroll
    for (int tn = 0; tn < 4; ++tn) {
      const int n = n0 + wn * 64 + tn * 16 + lrow;
      const float bv = Bias[n];
      #pragma unroll
      for (int r = 0; r < 8; ++r)
        Y[(size_t)(mb + r) * N_TOT + n] = acc[tm][tn][r] + bv;
    }
  }
}

// ===========================================================================
extern "C" void kernel_launch(void* const* d_in, const int* in_sizes, int n_in,
                              void* d_out, int out_size, void* d_ws, size_t ws_size,
                              hipStream_t stream) {
  const float* X    = (const float*)d_in[0];   // (4, 4096, 1024) f32
  const float* W    = (const float*)d_in[1];   // (1024, 4096)    f32
  const float* bias = (const float*)d_in[2];   // (4096,)         f32
  float*       Y    = (float*)d_out;           // (4, 4096, 4096) f32

  dim3 grid(N_TOT / BN, M_TOT / BM);           // 32 x 128 blocks

  if (ws_size >= WS_NEED) {
    unsigned short* Xh  = (unsigned short*)d_ws;
    unsigned short* Xl  = Xh + XH_ELEMS;
    unsigned short* Wth = Xl + XH_ELEMS;
    unsigned short* Wtl = Wth + WT_ELEMS;

    const int n8 = (int)(XH_ELEMS / 8);        // 2,097,152
    split_x_kernel<<<dim3((n8 + 255) / 256), 256, 0, stream>>>(X, Xh, Xl, n8);
    split_wt_kernel<<<dim3(K_TOT / 64, N_TOT / 64), 256, 0, stream>>>(W, Wth, Wtl);
    qdense_gemm_wmma<<<grid, 256, 0, stream>>>(Xh, Xl, Wth, Wtl, bias, Y);
  } else {
    qdense_fused_wmma<<<grid, 256, 0, stream>>>(X, W, bias, Y);
  }
}
